// MurrayLawModule_85856396247188
// MI455X (gfx1250) — matrix operand order repaired and checked by reading, not compiled, not run on hardware
//
#include <hip/hip_runtime.h>
#include <math.h>

typedef __attribute__((ext_vector_type(16))) _Float16 v16h;
typedef __attribute__((ext_vector_type(8)))  _Float16 v8h;
typedef __attribute__((ext_vector_type(8)))  float    v8f;

#define NN 300000
#define NE 3000000
#define NTILES (NN / 16)

// ---- dynamic LDS layout (bytes) ----
#define OFF_WT1RC 0          // [64][192] f16 = 24576 B  (rc_w1^T)
#define OFF_WT1CP 24576      // [64][192] f16 = 24576 B  (cp_w1^T)
#define OFF_WT2   49152      // [32][64]  f16 = 4096 B   (rc_w2^T)
#define OFF_B1RC  53248      // 64 f32
#define OFF_B2RC  53504      // 32 f32
#define OFF_W3    53632      // 32 f32
#define OFF_B1CP  53760      // 64 f32
#define OFF_W2CP  54016      // 64 f32
#define OFF_SCAL  54272      // 2 f32 (rc_b3, cp_b2) + pad
#define OFF_WAVE  54528      // per-wave regions start (32B aligned)
#define WAVE_STRIDE 6272     // h1f16 2048 + scratch f32 4096 + gate 64 + pad
#define SMEM_TOTAL (OFF_WAVE + 8 * WAVE_STRIDE)   // 104704 B

// ---------------- edge kernels: two-smallest-distinct-neighbor ----------------
__global__ void edge_pass1(const int* __restrict__ e0, const int* __restrict__ e1,
                           int* __restrict__ min1, int nE) {
  int i = blockIdx.x * blockDim.x + threadIdx.x;
  if (i >= nE) return;
  int a = e0[i], b = e1[i];
  if (a != b) { atomicMin(&min1[a], b); atomicMin(&min1[b], a); }
}

__global__ void edge_pass2(const int* __restrict__ e0, const int* __restrict__ e1,
                           const int* __restrict__ min1, int* __restrict__ min2, int nE) {
  int i = blockIdx.x * blockDim.x + threadIdx.x;
  if (i >= nE) return;
  int a = e0[i], b = e1[i];
  if (a != b) {
    if (b != min1[a]) atomicMin(&min2[a], b);
    if (a != min1[b]) atomicMin(&min2[b], a);
  }
}

// ---------------- main fused WMMA kernel ----------------
__global__ __launch_bounds__(256, 1) void murray_main(
    const float* __restrict__ nf, const float* __restrict__ radii,
    const float* __restrict__ rc_w1, const float* __restrict__ rc_b1,
    const float* __restrict__ rc_w2, const float* __restrict__ rc_b2,
    const float* __restrict__ rc_w3, const float* __restrict__ rc_b3,
    const float* __restrict__ cp_w1, const float* __restrict__ cp_b1,
    const float* __restrict__ cp_w2, const float* __restrict__ cp_b2,
    const int* __restrict__ types,
    const int* __restrict__ min1, const int* __restrict__ min2,
    float* __restrict__ out, float* __restrict__ sums)
{
  extern __shared__ char smem[];
  _Float16* wt1rc = (_Float16*)(smem + OFF_WT1RC);
  _Float16* wt1cp = (_Float16*)(smem + OFF_WT1CP);
  _Float16* wt2   = (_Float16*)(smem + OFF_WT2);
  float* b1rc = (float*)(smem + OFF_B1RC);
  float* b2rc = (float*)(smem + OFF_B2RC);
  float* w3   = (float*)(smem + OFF_W3);
  float* b1cp = (float*)(smem + OFF_B1CP);
  float* w2cp = (float*)(smem + OFF_W2CP);
  float* scal = (float*)(smem + OFF_SCAL);

  const int tid = threadIdx.x;

  // Stage weights transposed to f16 [N][K] (K contiguous per output column).
  for (int idx = tid; idx < 64 * 192; idx += 256) {
    int n = idx / 192, k = idx - n * 192;
    wt1rc[idx] = (_Float16)rc_w1[k * 64 + n];
    wt1cp[idx] = (_Float16)cp_w1[k * 64 + n];
  }
  for (int idx = tid; idx < 32 * 64; idx += 256) {
    int n = idx >> 6, k = idx & 63;
    wt2[idx] = (_Float16)rc_w2[k * 32 + n];
  }
  if (tid < 64) { b1cp[tid] = cp_b1[tid]; w2cp[tid] = cp_w2[tid]; b1rc[tid] = rc_b1[tid]; }
  if (tid < 32) { b2rc[tid] = rc_b2[tid]; w3[tid] = rc_w3[tid]; }
  if (tid == 0) { scal[0] = rc_b3[0]; scal[1] = cp_b2[0]; }
  __syncthreads();

  const int wave = tid >> 5;
  const int lane = tid & 31;
  const int r = lane & 15;       // A row / B column within tile
  const int h = lane >> 4;       // half-wave selector per CDNA5 WMMA layouts

  char* wbase = smem + OFF_WAVE + wave * WAVE_STRIDE;
  _Float16* h1 = (_Float16*)wbase;                 // [16][64] f16
  float* scratch = (float*)(wbase + 2048);         // [16][64] f32 (reused)
  float* gateL = (float*)(wbase + 2048 + 4096);    // [16] f32

  const int wid = blockIdx.x * 8 + wave;
  const int wstride = gridDim.x * 8;

  for (int tile = wid; tile < NTILES; tile += wstride) {
    const int tb = tile * 16;
    const int m = tb + r;
    const int m1 = min1[m], m2v = min2[m];
    const int c1 = m1 < NN ? m1 : NN - 1;
    const int c2 = m2v < NN ? m2v : NN - 1;
    const int validf = (m2v < NN);

    const float* rowp[3];
    rowp[0] = nf + (size_t)m * 64;
    rowp[1] = nf + (size_t)c1 * 64;
    rowp[2] = nf + (size_t)c2 * 64;

    // Build A fragments (16x32 f16 per K-step) straight from global.
    // Lane(r,h) holds row r, K chunk [32s+16h, +16).
    v16h afr[6];
#pragma unroll
    for (int s = 0; s < 6; ++s) {
      const float* rp = rowp[s >> 1];
      const int kb = (s & 1) * 32;
      const float4* p0 = (const float4*)(rp + kb + 16 * h);
      const float4* p1 = (const float4*)(rp + kb + 16 * h + 8);
      float4 u0 = p0[0], u1 = p0[1], u2 = p1[0], u3 = p1[1];
      v16h a;
      a[0] = (_Float16)u0.x;  a[1] = (_Float16)u0.y;  a[2] = (_Float16)u0.z;  a[3] = (_Float16)u0.w;
      a[4] = (_Float16)u1.x;  a[5] = (_Float16)u1.y;  a[6] = (_Float16)u1.z;  a[7] = (_Float16)u1.w;
      a[8] = (_Float16)u2.x;  a[9] = (_Float16)u2.y;  a[10] = (_Float16)u2.z; a[11] = (_Float16)u2.w;
      a[12] = (_Float16)u3.x; a[13] = (_Float16)u3.y; a[14] = (_Float16)u3.z; a[15] = (_Float16)u3.w;
      afr[s] = a;
    }

    // Layer 1 for both MLPs: D[16x64] = A[16x192] x W[192x64]
#pragma unroll
    for (int nt = 0; nt < 4; ++nt) {
      v8f accR = {}, accC = {};
      const int col = nt * 16 + r;   // output column handled by this lane
#pragma unroll
      for (int s = 0; s < 6; ++s) {
        v16h bR = *(const v16h*)(wt1rc + col * 192 + s * 32 + 16 * h);
        accR = __builtin_amdgcn_wmma_f32_16x16x32_f16(false, afr[s], false, bR,
                                                      (short)0, accR, false, false);
      }
#pragma unroll
      for (int s = 0; s < 6; ++s) {
        v16h bC = *(const v16h*)(wt1cp + col * 192 + s * 32 + 16 * h);
        accC = __builtin_amdgcn_wmma_f32_16x16x32_f16(false, afr[s], false, bC,
                                                      (short)0, accC, false, false);
      }
      const float brc = b1rc[col], bcp = b1cp[col];
#pragma unroll
      for (int i = 0; i < 8; ++i) {          // D vgpr i -> row i+8h, col
        const int row = i + 8 * h;
        float hr = accR[i] + brc; hr = hr > 0.0f ? hr : 0.0f;
        h1[row * 64 + col] = (_Float16)hr;
        float hc = accC[i] + bcp; hc = hc > 0.0f ? hc : 0.0f;
        scratch[row * 64 + col] = hc;
      }
    }

    // Compliance: hc[16x64] @ cp_w2 + b -> sigmoid (lanes 0-15, one row each)
    float compliance = 0.0f;
    if (lane < 16) {
      float s = scal[1];
#pragma unroll 8
      for (int j = 0; j < 64; ++j) s += scratch[lane * 64 + j] * w2cp[j];
      compliance = 1.0f / (1.0f + __expf(-s));
    }

    // Layer 2 (rc): h2[16x32] = h1[16x64] x rc_w2[64x32]
    v16h a2[2];
#pragma unroll
    for (int s2 = 0; s2 < 2; ++s2) {
      v8h lo = *(const v8h*)(h1 + r * 64 + s2 * 32 + 16 * h);
      v8h hi = *(const v8h*)(h1 + r * 64 + s2 * 32 + 16 * h + 8);
      a2[s2] = __builtin_shufflevector(lo, hi, 0, 1, 2, 3, 4, 5, 6, 7,
                                       8, 9, 10, 11, 12, 13, 14, 15);
    }
#pragma unroll
    for (int nt2 = 0; nt2 < 2; ++nt2) {
      v8f acc2 = {};
      const int col2 = nt2 * 16 + r;
#pragma unroll
      for (int s2 = 0; s2 < 2; ++s2) {
        v16h b2 = *(const v16h*)(wt2 + col2 * 64 + s2 * 32 + 16 * h);
        acc2 = __builtin_amdgcn_wmma_f32_16x16x32_f16(false, a2[s2], false, b2,
                                                      (short)0, acc2, false, false);
      }
      const float bb = b2rc[col2];
#pragma unroll
      for (int i = 0; i < 8; ++i) {
        float v = acc2[i] + bb; v = v > 0.0f ? v : 0.0f;
        scratch[(i + 8 * h) * 32 + col2] = v;  // reuse scratch (wave DS is in-order)
      }
    }

    // corr = h2 @ rc_w3 + b3; violation + gating (lanes 0-15)
    float viol = 0.0f; int active = 0;
    if (lane < 16) {
      float s = scal[0];
#pragma unroll 8
      for (int j = 0; j < 32; ++j) s += scratch[lane * 32 + j] * w3[j];
      const float pr = radii[m];
      const float r1 = radii[c1], r2 = radii[c2];
      const float pr3 = pr * pr * pr;
      viol = fabsf(r1 * r1 * r1 + r2 * r2 * r2 - pr3) / pr3;
      active = (types[m] == 1) && validf;
      const int upd = active && (viol > 0.2f);
      gateL[lane] = upd ? s : 0.0f;
    }

    float vv = (lane < 16 && active) ? viol : 0.0f;
    float cc = (lane < 16 && active) ? compliance : 0.0f;
    float cn = (lane < 16 && active) ? 1.0f : 0.0f;
#pragma unroll
    for (int off = 16; off > 0; off >>= 1) {
      vv += __shfl_xor(vv, off, 32);
      cc += __shfl_xor(cc, off, 32);
      cn += __shfl_xor(cn, off, 32);
    }
    if (lane == 0 && cn > 0.0f) {
      atomicAdd(&sums[0], vv);
      atomicAdd(&sums[1], cc);
      atomicAdd(&sums[2], cn);
    }

    // updated = f + 0.1*gate*tanh(f) : 2 lanes per row, 32 floats each
    {
      const int row = lane >> 1, hs = lane & 1;
      const int mo = tb + row;
      const float g = gateL[row] * 0.1f;
      const float* fp = nf + (size_t)mo * 64 + hs * 32;
      float* op = out + (size_t)mo * 64 + hs * 32;
#pragma unroll
      for (int k = 0; k < 8; ++k) {
        float4 f = ((const float4*)fp)[k];
        float4 o;
        o.x = f.x + g * tanhf(f.x);
        o.y = f.y + g * tanhf(f.y);
        o.z = f.z + g * tanhf(f.z);
        o.w = f.w + g * tanhf(f.w);
        ((float4*)op)[k] = o;
      }
    }
  }
}

__global__ void finalize_kernel(const float* __restrict__ sums, float* __restrict__ out) {
  const float cnt = fmaxf(sums[2], 1.0f);
  out[(size_t)NN * 64 + 0] = sums[0] / cnt;
  out[(size_t)NN * 64 + 1] = sums[1] / cnt;
}

extern "C" void kernel_launch(void* const* d_in, const int* in_sizes, int n_in,
                              void* d_out, int out_size, void* d_ws, size_t ws_size,
                              hipStream_t stream) {
  const float* nf    = (const float*)d_in[0];
  const float* radii = (const float*)d_in[1];
  const float* rc_w1 = (const float*)d_in[2];
  const float* rc_b1 = (const float*)d_in[3];
  const float* rc_w2 = (const float*)d_in[4];
  const float* rc_b2 = (const float*)d_in[5];
  const float* rc_w3 = (const float*)d_in[6];
  const float* rc_b3 = (const float*)d_in[7];
  const float* cp_w1 = (const float*)d_in[8];
  const float* cp_b1 = (const float*)d_in[9];
  const float* cp_w2 = (const float*)d_in[10];
  const float* cp_b2 = (const float*)d_in[11];
  const int* eidx    = (const int*)d_in[12];
  const int* types   = (const int*)d_in[13];

  int* min1 = (int*)d_ws;
  int* min2 = min1 + NN;
  float* sums = (float*)(min2 + NN);
  float* out = (float*)d_out;

  // Sentinel 0x7F7F7F7F > any node id; positive for signed atomicMin.
  hipMemsetAsync(min1, 0x7F, (size_t)2 * NN * sizeof(int), stream);
  hipMemsetAsync(sums, 0, 4 * sizeof(float), stream);

  const int* e0 = eidx;
  const int* e1 = eidx + NE;
  const int eb = (NE + 255) / 256;
  edge_pass1<<<eb, 256, 0, stream>>>(e0, e1, min1, NE);
  edge_pass2<<<eb, 256, 0, stream>>>(e0, e1, min1, min2, NE);

  static bool attr_set = false;
  if (!attr_set) {
    hipFuncSetAttribute((const void*)murray_main,
                        hipFuncAttributeMaxDynamicSharedMemorySize, SMEM_TOTAL);
    attr_set = true;
  }
  murray_main<<<512, 256, SMEM_TOTAL, stream>>>(
      nf, radii, rc_w1, rc_b1, rc_w2, rc_b2, rc_w3, rc_b3,
      cp_w1, cp_b1, cp_w2, cp_b2, types, min1, min2, out, sums);
  finalize_kernel<<<1, 1, 0, stream>>>(sums, out);
}
